// DynamicGatingNetwork_29205777613023
// MI455X (gfx1250) — matrix operand order repaired and checked by reading, not compiled, not run on hardware
//
#include <hip/hip_runtime.h>
#include <hip/hip_bf16.h>
#include <stdint.h>

// MI455X (gfx1250): wave32, WMMA 16x16x32 bf16.
// Path A (preferred, needs d_ws): W axis split into up to 8 segments per batch
//   -> up to 1024 workgroups for TLP/latency hiding; partial Gram + sums in ws,
//   reduction kernel finishes std/threshold/count.
// Path B (fallback): fused single kernel, one workgroup per batch.

typedef __attribute__((ext_vector_type(16))) __bf16      v16bf;
typedef __attribute__((ext_vector_type(8)))  float       v8f;
typedef __attribute__((ext_vector_type(8)))  unsigned int v8u;

#define B_BATCH 128
#define W_LEN   8192
#define D_DIM   64
#define CHUNK   64      // w-rows staged per iteration (= 32 bf16x2 pairs)
#define PPAD    33      // 32 pairs + 1 dword pad -> conflict-free LDS pitch

__device__ __forceinline__ unsigned int pack_bf16x2(float f0, float f1) {
    // round-half-up to bf16, pack (f0 -> lo, f1 -> hi) with a single v_perm_b32
    const unsigned int u0 = __builtin_bit_cast(unsigned int, f0) + 0x8000u;
    const unsigned int u1 = __builtin_bit_cast(unsigned int, f1) + 0x8000u;
    return __builtin_amdgcn_perm(u1, u0, 0x07060302u);
}

// Workgroup barrier that drains ONLY the LDS counter (DScnt), not in-flight
// global prefetch loads (__syncthreads would force s_wait_loadcnt 0 too and
// kill the depth-1 software pipeline). Split-barrier per CDNA5 ISA ch.5/6.
__device__ __forceinline__ void wg_barrier_lds() {
    asm volatile("s_wait_dscnt 0x0\n\t"
                 "s_barrier_signal -1\n\t"
                 "s_barrier_wait -1" ::: "memory");
}

// Stage one 64-row chunk (held in registers) into LDS and run both K=32 WMMAs.
__device__ __forceinline__ void gram_step(unsigned int (&chunk)[D_DIM][PPAD],
                                          const float (&c0)[4], const float (&c1)[4],
                                          int c, int pbase,
                                          int ti, int tj, int ln, int half,
                                          v8f& acc, float& sacc) {
    #pragma unroll
    for (int it = 0; it < 4; ++it) {
        sacc += c0[it] + c1[it];
        chunk[c][pbase + it * 8] = pack_bf16x2(c0[it], c1[it]);
    }
    wg_barrier_lds();                       // producers' ds_stores -> visible
    #pragma unroll
    for (int kk = 0; kk < 2; ++kk) {
        v8u av, bv;
        #pragma unroll
        for (int r = 0; r < 8; ++r) {
            // A (16x32, M=i-tile rows, K=w) / B (32x16, K=w, N=j-tile cols)
            const int pA = kk * 16 + half * 8 + r;
            const int pB = kk * 16 + half * 8 + r;
            av[r] = chunk[ti * 16 + ln][pA];
            bv[r] = chunk[tj * 16 + ln][pB];
        }
        acc = __builtin_amdgcn_wmma_f32_16x16x32_bf16(
            false, __builtin_bit_cast(v16bf, av),
            false, __builtin_bit_cast(v16bf, bv),
            (short)0, acc, false, false);
    }
    wg_barrier_lds();                       // consumers done -> safe to overwrite
}

// Stream x[w_begin:w_end) of one batch. Single per-thread pointer bumped by a
// constant stride; all 8 chunk loads are non-temporal (stream > L2) and use
// immediate IOFFSETs (one s_clause). Depth-1 register prefetch; last chunk
// peeled so the hot loop has no guard.
__device__ __forceinline__ void gram_loop(const float* __restrict__ xb,
                                          int w_begin, int w_end,
                                          unsigned int (&chunk)[D_DIM][PPAD],
                                          int c, int pbase,
                                          int ti, int tj, int ln, int half,
                                          v8f& acc, float& sacc) {
    const float* ptr = xb + (size_t)w_begin * D_DIM + pbase * 2 * D_DIM + c;
    float cur0[4], cur1[4];
    #pragma unroll
    for (int it = 0; it < 4; ++it) {
        cur0[it] = __builtin_nontemporal_load(&ptr[it * 16 * D_DIM]);
        cur1[it] = __builtin_nontemporal_load(&ptr[it * 16 * D_DIM + D_DIM]);
    }
    const int nch = (w_end - w_begin) / CHUNK;
    for (int ch = 0; ch < nch - 1; ++ch) {
        ptr += CHUNK * D_DIM;                       // one 64-bit add per chunk
        float nxt0[4], nxt1[4];
        #pragma unroll
        for (int it = 0; it < 4; ++it) {            // issued before barriers
            nxt0[it] = __builtin_nontemporal_load(&ptr[it * 16 * D_DIM]);
            nxt1[it] = __builtin_nontemporal_load(&ptr[it * 16 * D_DIM + D_DIM]);
        }
        gram_step(chunk, cur0, cur1, c, pbase, ti, tj, ln, half, acc, sacc);
        #pragma unroll
        for (int it = 0; it < 4; ++it) { cur0[it] = nxt0[it]; cur1[it] = nxt1[it]; }
    }
    gram_step(chunk, cur0, cur1, c, pbase, ti, tj, ln, half, acc, sacc);
}

// ---------------- Path A, kernel 1: partial Gram per (batch, segment) --------
__launch_bounds__(512)
__global__ void gram_partial_kernel(const float* __restrict__ x,
                                    float* __restrict__ pG,   // [B*nseg][64*64]
                                    float* __restrict__ pS,   // [B*nseg][64]
                                    int nseg) {
    __shared__ unsigned int chunk[D_DIM][PPAD];
    __shared__ float psum[8][D_DIM];

    const int t    = threadIdx.x;
    const int lane = t & 31;
    const int half = (lane >> 4) & 1;
    const int ln   = lane & 15;
    const int wave = t >> 5;
    const int ti   = wave >> 2;
    const int tj   = wave & 3;

    const int blk  = blockIdx.x;
    const int b    = blk / nseg;
    const int seg  = blk - b * nseg;
    const int wseg = W_LEN / nseg;

    const float* xb = x + (size_t)b * W_LEN * D_DIM;
    const int c     = t & 63;
    const int pbase = t >> 6;

    float sacc = 0.0f;
    v8f acc = {};
    gram_loop(xb, seg * wseg, (seg + 1) * wseg, chunk, c, pbase, ti, tj, ln, half,
              acc, sacc);

    // deterministic column-sum reduction, write partial sums
    psum[pbase][c] = sacc;
    __syncthreads();
    if (t < D_DIM) {
        float s = 0.0f;
        #pragma unroll
        for (int g = 0; g < 8; ++g) s += psum[g][t];
        pS[(size_t)blk * D_DIM + t] = s;
    }
    // write partial Gram tile
    float* g = pG + (size_t)blk * (D_DIM * D_DIM);
    #pragma unroll
    for (int r = 0; r < 8; ++r) {
        const int gm = ti * 16 + r + 8 * half;
        const int gn = tj * 16 + ln;
        g[gm * D_DIM + gn] = acc[r];
    }
}

// ---------------- Path A, kernel 2: reduce segments + threshold count --------
__launch_bounds__(256)
__global__ void corr_reduce_kernel(const float* __restrict__ pG,
                                   const float* __restrict__ pS,
                                   float* __restrict__ out, int nseg) {
    __shared__ float ssum[D_DIM];
    __shared__ float gdiag[D_DIM];
    __shared__ float stdE[D_DIM];
    __shared__ int   cnt[D_DIM];

    const int t = threadIdx.x;
    const int b = blockIdx.x;

    if (t < D_DIM) {
        float s = 0.0f;
        for (int k = 0; k < nseg; ++k)
            s += pS[(size_t)(b * nseg + k) * D_DIM + t];
        ssum[t] = s;
        cnt[t]  = 0;
    }
    // each thread owns 16 consecutive Gram entries of one row
    const int row = t >> 2;
    const int c0  = (t & 3) * 16;
    float gv[16];
    #pragma unroll
    for (int q = 0; q < 16; ++q) gv[q] = 0.0f;
    for (int k = 0; k < nseg; ++k) {
        const float* g = pG + (size_t)(b * nseg + k) * (D_DIM * D_DIM)
                         + row * D_DIM + c0;
        #pragma unroll
        for (int q = 0; q < 16; ++q) gv[q] += g[q];
    }
    if (row >= c0 && row < c0 + 16) gdiag[row] = gv[row - c0];
    __syncthreads();

    const float invW   = 1.0f / (float)W_LEN;
    const float invWm1 = 1.0f / (float)(W_LEN - 1);
    if (t < D_DIM) {
        const float var = (gdiag[t] - ssum[t] * ssum[t] * invW) * invWm1;
        stdE[t] = sqrtf(fmaxf(var, 0.0f)) + 1e-8f;
    }
    __syncthreads();

    int local = 0;
    const float sr = stdE[row];
    const float mr = ssum[row];
    #pragma unroll
    for (int q = 0; q < 16; ++q) {
        const int col = c0 + q;
        const float cov = (gv[q] - mr * ssum[col] * invW) * invWm1;
        if (fabsf(cov) > 0.3f * sr * stdE[col]) ++local;
    }
    atomicAdd(&cnt[row], local);
    __syncthreads();
    if (t < D_DIM) out[(size_t)b * D_DIM + t] = (float)cnt[t] - 1.0f;
}

// ---------------- Path B: fused fallback (no workspace needed) ---------------
__launch_bounds__(512)
__global__ void corr_fused_kernel(const float* __restrict__ x,
                                  float* __restrict__ out) {
    __shared__ unsigned int chunk[D_DIM][PPAD];
    __shared__ float psum[8][D_DIM];
    __shared__ float ssum[D_DIM];
    __shared__ float gdiag[D_DIM];
    __shared__ float stdE[D_DIM];
    __shared__ int   cnt[D_DIM];

    const int t    = threadIdx.x;
    const int lane = t & 31;
    const int half = (lane >> 4) & 1;
    const int ln   = lane & 15;
    const int wave = t >> 5;
    const int ti   = wave >> 2;
    const int tj   = wave & 3;
    const int b    = blockIdx.x;

    const float* xb = x + (size_t)b * W_LEN * D_DIM;
    if (t < D_DIM) cnt[t] = 0;

    const int c     = t & 63;
    const int pbase = t >> 6;

    float sacc = 0.0f;
    v8f acc = {};
    gram_loop(xb, 0, W_LEN, chunk, c, pbase, ti, tj, ln, half, acc, sacc);

    psum[pbase][c] = sacc;
    __syncthreads();
    if (t < D_DIM) {
        float s = 0.0f;
        #pragma unroll
        for (int g = 0; g < 8; ++g) s += psum[g][t];
        ssum[t] = s;
    }
    __syncthreads();

    if (ti == tj) {
        #pragma unroll
        for (int r = 0; r < 8; ++r) {
            // C/D layout: vgpr r holds (M=r, N=lane) lanes 0-15,
            // (M=r+8, N=lane-16) lanes 16-31 -> diag lane = r + 24*half
            if (lane == r + 24 * half) gdiag[ti * 16 + r + 8 * half] = acc[r];
        }
    }
    __syncthreads();

    const float invW   = 1.0f / (float)W_LEN;
    const float invWm1 = 1.0f / (float)(W_LEN - 1);
    if (t < D_DIM) {
        const float var = (gdiag[t] - ssum[t] * ssum[t] * invW) * invWm1;
        stdE[t] = sqrtf(fmaxf(var, 0.0f)) + 1e-8f;
    }
    __syncthreads();

    #pragma unroll
    for (int r = 0; r < 8; ++r) {
        const int gm = ti * 16 + r + 8 * half;
        const int gn = tj * 16 + ln;
        const float cov = (acc[r] - ssum[gm] * ssum[gn] * invW) * invWm1;
        if (fabsf(cov) > 0.3f * stdE[gm] * stdE[gn]) atomicAdd(&cnt[gm], 1);
    }
    __syncthreads();
    if (t < D_DIM) out[(size_t)b * D_DIM + t] = (float)cnt[t] - 1.0f;
}

extern "C" void kernel_launch(void* const* d_in, const int* in_sizes, int n_in,
                              void* d_out, int out_size, void* d_ws, size_t ws_size,
                              hipStream_t stream) {
    (void)in_sizes; (void)n_in; (void)out_size;
    const float* x = (const float*)d_in[0];
    float* out = (float*)d_out;

    // choose the largest segment split the workspace allows (deterministic:
    // depends only on ws_size, which is fixed across calls)
    const size_t per_blk = (size_t)(D_DIM * D_DIM + D_DIM) * sizeof(float);
    int nseg = 0;
    for (int s = 8; s >= 1; s >>= 1) {
        if ((size_t)B_BATCH * s * per_blk <= ws_size) { nseg = s; break; }
    }

    if (nseg > 0) {
        float* pG = (float*)d_ws;                                   // B*nseg*4096
        float* pS = pG + (size_t)B_BATCH * nseg * (D_DIM * D_DIM);  // B*nseg*64
        gram_partial_kernel<<<dim3(B_BATCH * nseg), dim3(512), 0, stream>>>(
            x, pG, pS, nseg);
        corr_reduce_kernel<<<dim3(B_BATCH), dim3(256), 0, stream>>>(
            pG, pS, out, nseg);
    } else {
        corr_fused_kernel<<<dim3(B_BATCH), dim3(512), 0, stream>>>(x, out);
    }
}
